// GPT_50689204027413
// MI455X (gfx1250) — matrix operand order repaired and checked by reading, not compiled, not run on hardware
//
#include <hip/hip_runtime.h>
#include <math.h>

typedef _Float16 h16;
typedef __attribute__((ext_vector_type(16))) _Float16 v16h;
typedef __attribute__((ext_vector_type(8)))  float    v8f;

#define WMMA_F16(A, B, C) \
  __builtin_amdgcn_wmma_f32_16x16x32_f16(false, (A), false, (B), (short)0, (C), false, false)

// Async global->LDS copy (gfx1250 ASYNCcnt path). LDS dest is the low 32 bits
// of the generic pointer (LDS aperture maps by truncation to addr[31:0]).
__device__ __forceinline__ void async_g2l_b64(const void* gsrc, void* lds_dst) {
  unsigned l = (unsigned)(unsigned long long)lds_dst;
  unsigned long long g = (unsigned long long)gsrc;
  asm volatile("global_load_async_to_lds_b64 %0, %1, off" :: "v"(l), "v"(g) : "memory");
}
__device__ __forceinline__ void wait_async0() {
  asm volatile("s_wait_asynccnt 0x0" ::: "memory");
}

// Map element (row, k) of a row-major [M x K] f16 matrix into the blocked
// A-fragment layout used by this file's WMMA (contiguous k order): block = (16 rows x 32 K), 32 lanes x 16 halves.
// Lane = g*16 + row%16 ; half j holds K = 16*g + j  (inverted here).
__device__ __forceinline__ size_t swz_idx(int row, int k, int KC) {
  int g = (k >> 4) & 1;
  int j = k & 15;
  return ((size_t)((row >> 4) * KC + (k >> 5)) * 32 + g * 16 + (row & 15)) * 16 + j;
}

// ---------------------------------------------------------------------------
// Elementwise / transpose helpers
// ---------------------------------------------------------------------------

// x = tok_emb[tokens] + pos_emb ; f32 linear + f16 swizzled copy
__global__ void embed_kernel(const int* __restrict__ tok,
                             const float* __restrict__ te,
                             const float* __restrict__ pe,
                             float* __restrict__ x, h16* __restrict__ xh) {
  int i = blockIdx.x * 256 + threadIdx.x;   // over 4096*1024
  int t = i >> 10, e = i & 1023;
  int s = t & 1023;
  float v = te[(size_t)tok[t] * 1024 + e] + pe[(size_t)s * 1024 + e];
  x[i] = v;
  xh[swz_idx(t, e, 32)] = (h16)v;
}

// wT[n*K + k] = (f16) w[k*N + n]. Read side: async global->LDS bulk copy
// (one 32x32 f32 tile, padded rows), then coalesced transposed f16 writes.
__global__ __launch_bounds__(256) void convT_kernel(const float* __restrict__ w,
                                                    h16* __restrict__ wT,
                                                    int K, int N) {
  __shared__ float tile[32 * 34];           // row stride 34 floats (136B, 8B-aligned)
  int nt = N >> 5;
  int bn = blockIdx.x % nt, bk = blockIdx.x / nt;
  int tid = threadIdx.x;

  // async copy: each thread moves 2 x 8 bytes of its row
  int row = tid >> 3;                        // 0..31  (k-local)
  int s8  = tid & 7;                         // 8B segment index (row = 16 segs)
  const float* gr = w + (size_t)(bk * 32 + row) * N + bn * 32;
#pragma unroll
  for (int it = 0; it < 2; ++it) {
    int seg = s8 + 8 * it;                   // 0..15
    async_g2l_b64(gr + seg * 2, &tile[row * 34 + seg * 2]);
  }
  wait_async0();
  __syncthreads();

  int tx = tid & 31, ty = tid >> 5;
#pragma unroll
  for (int r = 0; r < 4; ++r) {
    int nn = bn * 32 + ty + r * 8;
    wT[(size_t)nn * K + bk * 32 + tx] = (h16)tile[tx * 34 + (ty + r * 8)];
  }
}

// vT[(b*1024 + c)*1024 + s] = v[(b*1024 + s)*1024 + c] via LDS tile
__global__ __launch_bounds__(256) void vtrans_kernel(const h16* __restrict__ v,
                                                     h16* __restrict__ vT) {
  __shared__ h16 tile[32][33];
  int ct = blockIdx.x & 31;
  int st = (blockIdx.x >> 5) & 31;
  int b  = blockIdx.x >> 10;
  int tx = threadIdx.x & 31, ty = threadIdx.x >> 5;
#pragma unroll
  for (int r = 0; r < 4; ++r) {
    int s = st * 32 + ty + r * 8;
    tile[ty + r * 8][tx] = v[((size_t)(b * 1024 + s) << 10) + ct * 32 + tx];
  }
  __syncthreads();
#pragma unroll
  for (int r = 0; r < 4; ++r) {
    int c = ct * 32 + ty + r * 8;
    vT[((size_t)(b * 1024 + c) << 10) + st * 32 + tx] = tile[tx][ty + r * 8];
  }
}

// t = y + x (residual); LayerNorm(t)*g+b -> xout (f32 linear) and xh (f16 swizzled)
__global__ void resid_ln_kernel(const float* __restrict__ y,
                                const float* __restrict__ xin,
                                const float* __restrict__ gw,
                                const float* __restrict__ bw,
                                float* __restrict__ xout, h16* __restrict__ xh) {
  int t = blockIdx.x;
  int tid = threadIdx.x;
  __shared__ float sbuf[256];
  float local[4];
  float s = 0.f;
#pragma unroll
  for (int j = 0; j < 4; ++j) {
    int e = tid + j * 256;
    float v = y[(size_t)t * 1024 + e] + xin[(size_t)t * 1024 + e];
    local[j] = v;
    s += v;
  }
  sbuf[tid] = s; __syncthreads();
  for (int off = 128; off > 0; off >>= 1) {
    if (tid < off) sbuf[tid] += sbuf[tid + off];
    __syncthreads();
  }
  float mu = sbuf[0] * (1.f / 1024.f);
  __syncthreads();
  float s2 = 0.f;
#pragma unroll
  for (int j = 0; j < 4; ++j) { float d = local[j] - mu; s2 += d * d; }
  sbuf[tid] = s2; __syncthreads();
  for (int off = 128; off > 0; off >>= 1) {
    if (tid < off) sbuf[tid] += sbuf[tid + off];
    __syncthreads();
  }
  float rstd = rsqrtf(sbuf[0] * (1.f / 1024.f) + 1e-5f);
#pragma unroll
  for (int j = 0; j < 4; ++j) {
    int e = tid + j * 256;
    float o = (local[j] - mu) * rstd * gw[e] + bw[e];
    xout[(size_t)t * 1024 + e] = o;
    xh[swz_idx(t, e, 32)] = (h16)o;
  }
}

// ---------------------------------------------------------------------------
// WMMA GEMM:  out[M,N] = A(swizzled f16) @ BT[N,K](f16 = W^T rows) + bias
// One wave: 32x64 tile, 8 accumulators, double-buffered K loop + prefetch.
// OUT_MODE: 0 = f32 linear, 1 = f16 linear, 2 = f16 swizzled (for next GEMM A)
// ---------------------------------------------------------------------------
template <int OUT_MODE, int RELU>
__global__ __launch_bounds__(256) void gemm_wmma_kernel(
    const h16* __restrict__ ASwz, const h16* __restrict__ BT,
    const float* __restrict__ bias, void* __restrict__ outp,
    int M, int N, int K) {
  int wave = blockIdx.x * 8 + (threadIdx.x >> 5);
  int tilesN = N >> 6;
  int tilesM = M >> 5;              // 32-row tiles
  if (wave >= tilesM * tilesN) return;
  int tm = wave / tilesN, tn = wave - tm * tilesN;
  int lane = threadIdx.x & 31;
  int g = lane >> 4, n15 = lane & 15;
  int KC = K >> 5;

  const h16* a0p = ASwz + (size_t)(tm * 2 + 0) * KC * 512 + lane * 16;
  const h16* a1p = a0p + (size_t)KC * 512;
  const h16* bp0 = BT + (size_t)(tn * 64 + 0 * 16 + n15) * K + 16 * g;
  const h16* bp1 = BT + (size_t)(tn * 64 + 1 * 16 + n15) * K + 16 * g;
  const h16* bp2 = BT + (size_t)(tn * 64 + 2 * 16 + n15) * K + 16 * g;
  const h16* bp3 = BT + (size_t)(tn * 64 + 3 * 16 + n15) * K + 16 * g;

  v8f acc[2][4] = {};
  v16h a0 = *(const v16h*)(a0p);
  v16h a1 = *(const v16h*)(a1p);
  v16h b0 = *(const v16h*)(bp0);
  v16h b1 = *(const v16h*)(bp1);
  v16h b2 = *(const v16h*)(bp2);
  v16h b3 = *(const v16h*)(bp3);

  for (int ck = 1; ck < KC; ++ck) {
    v16h na0 = *(const v16h*)(a0p + (size_t)ck * 512);
    v16h na1 = *(const v16h*)(a1p + (size_t)ck * 512);
    v16h nb0 = *(const v16h*)(bp0 + ck * 32);
    v16h nb1 = *(const v16h*)(bp1 + ck * 32);
    v16h nb2 = *(const v16h*)(bp2 + ck * 32);
    v16h nb3 = *(const v16h*)(bp3 + ck * 32);
    if (ck + 1 < KC) {   // distance-2 prefetch -> global_prefetch_b8
      __builtin_prefetch(bp0 + (ck + 1) * 32, 0, 1);
      __builtin_prefetch(bp2 + (ck + 1) * 32, 0, 1);
      __builtin_prefetch(a0p + (size_t)(ck + 1) * 512, 0, 1);
      __builtin_prefetch(a1p + (size_t)(ck + 1) * 512, 0, 1);
    }
    acc[0][0] = WMMA_F16(a0, b0, acc[0][0]);
    acc[0][1] = WMMA_F16(a0, b1, acc[0][1]);
    acc[0][2] = WMMA_F16(a0, b2, acc[0][2]);
    acc[0][3] = WMMA_F16(a0, b3, acc[0][3]);
    acc[1][0] = WMMA_F16(a1, b0, acc[1][0]);
    acc[1][1] = WMMA_F16(a1, b1, acc[1][1]);
    acc[1][2] = WMMA_F16(a1, b2, acc[1][2]);
    acc[1][3] = WMMA_F16(a1, b3, acc[1][3]);
    a0 = na0; a1 = na1; b0 = nb0; b1 = nb1; b2 = nb2; b3 = nb3;
  }
  acc[0][0] = WMMA_F16(a0, b0, acc[0][0]);
  acc[0][1] = WMMA_F16(a0, b1, acc[0][1]);
  acc[0][2] = WMMA_F16(a0, b2, acc[0][2]);
  acc[0][3] = WMMA_F16(a0, b3, acc[0][3]);
  acc[1][0] = WMMA_F16(a1, b0, acc[1][0]);
  acc[1][1] = WMMA_F16(a1, b1, acc[1][1]);
  acc[1][2] = WMMA_F16(a1, b2, acc[1][2]);
  acc[1][3] = WMMA_F16(a1, b3, acc[1][3]);

  int KCn = N >> 5;
#pragma unroll
  for (int sub = 0; sub < 2; ++sub) {
#pragma unroll
    for (int nt = 0; nt < 4; ++nt) {
      int col = tn * 64 + nt * 16 + n15;
      float bv = bias[col];
#pragma unroll
      for (int r = 0; r < 8; ++r) {
        int row = tm * 32 + sub * 16 + r + 8 * g;
        float v = acc[sub][nt][r] + bv;
        if (RELU) v = fmaxf(v, 0.f);
        if (OUT_MODE == 0)      ((float*)outp)[(size_t)row * N + col] = v;
        else if (OUT_MODE == 1) ((h16*)outp)[(size_t)row * N + col] = (h16)v;
        else                    ((h16*)outp)[swz_idx(row, col, KCn)] = (h16)v;
      }
    }
  }
}

// ---------------------------------------------------------------------------
// Flash-style causal attention. One wave per (b, h, 16-query block).
// ---------------------------------------------------------------------------
__global__ __launch_bounds__(32) void attn_kernel(
    const h16* __restrict__ q_h, const h16* __restrict__ k_h,
    const h16* __restrict__ vT, h16* __restrict__ o_swz) {
  const int S = 1024, E = 1024;
  int idx = blockIdx.x;
  int qb = idx & 63;
  int h  = (idx >> 6) & 15;
  int b  = idx >> 10;
  int lane = threadIdx.x & 31;
  int g = lane >> 4, n15 = lane & 15;
  int qbase = qb * 16;

  __shared__ float sc[16][32];
  __shared__ h16   pb[16][32];
  __shared__ float rowf[16];

  v16h qa0, qa1;
  {
    const h16* qp = q_h + (size_t)(b * S + qbase + n15) * E + h * 64;
#pragma unroll
    for (int r = 0; r < 8; ++r) {
      int k0 = 2 * r + 16 * g;
      qa0[2 * r]     = (h16)((float)qp[k0] * 0.125f);
      qa0[2 * r + 1] = (h16)((float)qp[k0 + 1] * 0.125f);
      qa1[2 * r]     = (h16)((float)qp[32 + k0] * 0.125f);
      qa1[2 * r + 1] = (h16)((float)qp[32 + k0 + 1] * 0.125f);
    }
  }

  v8f acc[4] = {};
  float m_i = -__builtin_inff();
  float l_i = 0.f;

  const h16* kbase = k_h + (size_t)(b * S + n15) * E + h * 64 + 16 * g;
  const h16* vbase = vT + ((size_t)((b * 16 + h) * 64 + n15)) * S + 16 * g;

  int nch = (qbase + 16 + 31) >> 5;
  for (int ch = 0; ch < nch; ++ch) {
    int kb = ch * 32;
    if (ch + 1 < nch) {   // prefetch next chunk's K rows and V columns
      __builtin_prefetch(kbase + (size_t)(kb + 32) * E, 0, 1);
      __builtin_prefetch(kbase + (size_t)(kb + 48) * E, 0, 1);
      __builtin_prefetch(vbase + kb + 32, 0, 1);
    }
#pragma unroll
    for (int t = 0; t < 2; ++t) {
      int kbs = kb + 16 * t;
      const h16* kp = k_h + (size_t)(b * S + kbs + n15) * E + h * 64 + 16 * g;
      v16h kf0 = *(const v16h*)(kp);
      v16h kf1 = *(const v16h*)(kp + 32);
      v8f s = {};
      s = WMMA_F16(qa0, kf0, s);
      s = WMMA_F16(qa1, kf1, s);
      int kk = kbs + n15;
#pragma unroll
      for (int r = 0; r < 8; ++r) {
        int q = qbase + r + 8 * g;
        sc[r + 8 * g][16 * t + n15] = (kk <= q) ? s[r] : -__builtin_inff();
      }
    }
    __syncthreads();
    if (lane < 16) {
      float mx = m_i;
#pragma unroll
      for (int j = 0; j < 32; ++j) mx = fmaxf(mx, sc[lane][j]);
      float alpha = __expf(m_i - mx);
      float ls = 0.f;
#pragma unroll
      for (int j = 0; j < 32; ++j) {
        float p = __expf(sc[lane][j] - mx);
        ls += p;
        pb[lane][j] = (h16)p;
      }
      l_i = l_i * alpha + ls;
      m_i = mx;
      rowf[lane] = alpha;
    }
    __syncthreads();
#pragma unroll
    for (int r = 0; r < 8; ++r) {
      float al = rowf[r + 8 * g];
#pragma unroll
      for (int d = 0; d < 4; ++d) acc[d][r] *= al;
    }
    v16h pa;
#pragma unroll
    for (int r = 0; r < 8; ++r) {
      int k0 = 2 * r + 16 * g;
      pa[2 * r]     = pb[n15][k0];
      pa[2 * r + 1] = pb[n15][k0 + 1];
    }
#pragma unroll
    for (int d = 0; d < 4; ++d) {
      const h16* vp = vT + ((size_t)((b * 16 + h) * 64 + d * 16 + n15)) * S + kb + 16 * g;
      v16h vf = *(const v16h*)vp;
      acc[d] = WMMA_F16(pa, vf, acc[d]);
    }
    __syncthreads();
  }

  if (lane < 16) rowf[lane] = 1.0f / l_i;
  __syncthreads();
#pragma unroll
  for (int r = 0; r < 8; ++r) {
    float li = rowf[r + 8 * g];
    int row = b * S + qbase + r + 8 * g;
#pragma unroll
    for (int d = 0; d < 4; ++d) {
      int col = h * 64 + d * 16 + n15;
      o_swz[swz_idx(row, col, 32)] = (h16)(acc[d][r] * li);
    }
  }
}

// ---------------------------------------------------------------------------
// Host orchestration
// ---------------------------------------------------------------------------
extern "C" void kernel_launch(void* const* d_in, const int* in_sizes, int n_in,
                              void* d_out, int out_size, void* d_ws, size_t ws_size,
                              hipStream_t stream) {
  (void)in_sizes; (void)n_in; (void)out_size; (void)ws_size;
  const int*   tokens  = (const int*)  d_in[0];
  const float* tok_emb = (const float*)d_in[1];
  const float* pos_emb = (const float*)d_in[2];
  const float* wq = (const float*)d_in[3];
  const float* bq = (const float*)d_in[4];
  const float* wk = (const float*)d_in[5];
  const float* bk = (const float*)d_in[6];
  const float* wv = (const float*)d_in[7];
  const float* bv = (const float*)d_in[8];
  const float* wo = (const float*)d_in[9];
  const float* bo = (const float*)d_in[10];
  const float* w1 = (const float*)d_in[11];
  const float* b1 = (const float*)d_in[12];
  const float* w2 = (const float*)d_in[13];
  const float* b2 = (const float*)d_in[14];
  const float* ln1_g = (const float*)d_in[15];
  const float* ln1_b = (const float*)d_in[16];
  const float* ln2_g = (const float*)d_in[17];
  const float* ln2_b = (const float*)d_in[18];
  const float* w_out = (const float*)d_in[19];
  const float* b_out = (const float*)d_in[20];
  float* out = (float*)d_out;

  const int M = 4096;
  const int E = 1024, FF = 4096, V = 32000, L = 8;

  char* w = (char*)d_ws;
  float* x_f   = (float*)w; w += (size_t)M * E * 4;
  h16*   x_h   = (h16*)w;   w += (size_t)M * E * 2;   // swizzled
  h16*   q_h   = (h16*)w;   w += (size_t)M * E * 2;   // linear
  h16*   k_h   = (h16*)w;   w += (size_t)M * E * 2;   // linear
  h16*   v_h   = (h16*)w;   w += (size_t)M * E * 2;   // linear
  h16*   vT    = (h16*)w;   w += (size_t)M * E * 2;   // [b,c,s]
  h16*   a_h   = (h16*)w;   w += (size_t)M * E * 2;   // swizzled
  h16*   ff_h  = (h16*)w;   w += (size_t)M * FF * 2;  // swizzled
  float* ybuf  = (float*)w; w += (size_t)M * E * 4;
  h16*   wT    = (h16*)w;                             // weight staging (max 65.5 MB)

  dim3 blk256(256), blk32(32);
  int g_tok = (M * E) / 256;

  auto gemm_blocks = [](int m, int n) { return (((m >> 5) * (n >> 6)) + 7) / 8; };
  auto convT_blocks = [](int k, int n) { return (k >> 5) * (n >> 5); };

  embed_kernel<<<g_tok, blk256, 0, stream>>>(tokens, tok_emb, pos_emb, x_f, x_h);

  for (int i = 0; i < L; ++i) {
    const size_t oEE = (size_t)i * E * E, oE = (size_t)i * E;
    const size_t oEF = (size_t)i * E * FF;

    convT_kernel<<<convT_blocks(E, E), blk256, 0, stream>>>(wq + oEE, wT, E, E);
    gemm_wmma_kernel<1, 0><<<gemm_blocks(M, E), blk256, 0, stream>>>(x_h, wT, bq + oE, q_h, M, E, E);
    convT_kernel<<<convT_blocks(E, E), blk256, 0, stream>>>(wk + oEE, wT, E, E);
    gemm_wmma_kernel<1, 0><<<gemm_blocks(M, E), blk256, 0, stream>>>(x_h, wT, bk + oE, k_h, M, E, E);
    convT_kernel<<<convT_blocks(E, E), blk256, 0, stream>>>(wv + oEE, wT, E, E);
    gemm_wmma_kernel<1, 0><<<gemm_blocks(M, E), blk256, 0, stream>>>(x_h, wT, bv + oE, v_h, M, E, E);

    vtrans_kernel<<<4096, blk256, 0, stream>>>(v_h, vT);
    attn_kernel<<<4096, blk32, 0, stream>>>(q_h, k_h, vT, a_h);

    convT_kernel<<<convT_blocks(E, E), blk256, 0, stream>>>(wo + oEE, wT, E, E);
    gemm_wmma_kernel<0, 0><<<gemm_blocks(M, E), blk256, 0, stream>>>(a_h, wT, bo + oE, ybuf, M, E, E);
    resid_ln_kernel<<<M, blk256, 0, stream>>>(ybuf, x_f, ln1_g + oE, ln1_b + oE, x_f, x_h);

    convT_kernel<<<convT_blocks(E, FF), blk256, 0, stream>>>(w1 + oEF, wT, E, FF);
    gemm_wmma_kernel<2, 1><<<gemm_blocks(M, FF), blk256, 0, stream>>>(x_h, wT, b1 + (size_t)i * FF, ff_h, M, FF, E);
    convT_kernel<<<convT_blocks(FF, E), blk256, 0, stream>>>(w2 + oEF, wT, FF, E);
    gemm_wmma_kernel<0, 0><<<gemm_blocks(M, E), blk256, 0, stream>>>(ff_h, wT, b2 + oE, ybuf, M, E, FF);
    resid_ln_kernel<<<M, blk256, 0, stream>>>(ybuf, x_f, ln2_g + oE, ln2_b + oE, x_f, x_h);
  }

  convT_kernel<<<convT_blocks(E, V), blk256, 0, stream>>>(w_out, wT, E, V);
  gemm_wmma_kernel<0, 0><<<gemm_blocks(M, V), blk256, 0, stream>>>(x_h, wT, b_out, out, M, V, E);
}
